// DrBC_32289564131892
// MI455X (gfx1250) — compile-verified, matrix operand order checked by reading, not run on hardware
//
#include <hip/hip_runtime.h>
#include <math.h>

typedef __attribute__((ext_vector_type(16))) __bf16 v16bf;
typedef __attribute__((ext_vector_type(8)))  __bf16 v8bf;
typedef __attribute__((ext_vector_type(8)))  float  v8f;

// ---------------- helpers ----------------
__device__ __forceinline__ unsigned short f2bf(float f) {
    unsigned int u = __float_as_uint(f);
    u = (u + 0x7FFFu + ((u >> 16) & 1u)) >> 16;
    return (unsigned short)u;
}
__device__ __forceinline__ float bf2f(unsigned short b) {
    return __uint_as_float(((unsigned int)b) << 16);
}
// Load one 16x32 bf16 WMMA fragment chunk for this lane.
// p points at element (row, kb + half*8); second chunk is +16 elements.
__device__ __forceinline__ v16bf load_frag(const unsigned short* p) {
    v8bf lo = *(const v8bf*)(p);
    v8bf hi = *(const v8bf*)(p + 16);
    v16bf r;
    r.lo = lo;
    r.hi = hi;
    return r;
}
__device__ __forceinline__ float sigmoidf(float x) { return 1.f / (1.f + expf(-x)); }

// ---------------- trivial utility kernels ----------------
__global__ void zero_f32(float* p, int n) {
    int i = blockIdx.x * 256 + threadIdx.x;
    if (i < n) p[i] = 0.f;
}
__global__ void cast_f32_bf16(const float* in, unsigned short* out, int n) {
    int i = blockIdx.x * 256 + threadIdx.x;
    if (i < n) out[i] = f2bf(in[i]);
}

// ---------------- encoder: h0 = leaky_relu(x @ enc_w^T) ----------------
__global__ void enc_kernel(const float* __restrict__ x, const float* __restrict__ enc_w,
                           float* __restrict__ h0, unsigned short* __restrict__ h0bf, int total) {
    int i = blockIdx.x * 256 + threadIdx.x;
    if (i >= total) return;
    int n = i >> 7, d = i & 127;
    float acc = 0.f;
#pragma unroll
    for (int j = 0; j < 6; ++j) acc += x[n * 6 + j] * enc_w[d * 6 + j];
    float v = acc > 0.f ? acc : 0.01f * acc;
    h0[i] = v;
    h0bf[i] = f2bf(v);
}

// ---------------- edge aggregation: agg[dst] += norm * prev[src] ----------------
__global__ __launch_bounds__(256)
void edge_agg_kernel(const int* __restrict__ src, const int* __restrict__ dst,
                     const float* __restrict__ norm, const unsigned short* __restrict__ prev_bf,
                     float* __restrict__ agg, int E) {
    int e = blockIdx.x * 8 + (threadIdx.x >> 5);
    if (e >= E) return;
    int lane = threadIdx.x & 31;
    int s = src[e], d = dst[e];
    float w = norm[e];
    const unsigned short* p = prev_bf + (size_t)s * 128 + lane * 4;
    uint2 u = *(const uint2*)p;  // 4 bf16
    float v0 = bf2f((unsigned short)(u.x & 0xffff));
    float v1 = bf2f((unsigned short)(u.x >> 16));
    float v2 = bf2f((unsigned short)(u.y & 0xffff));
    float v3 = bf2f((unsigned short)(u.y >> 16));
    float* a = agg + (size_t)d * 128 + lane * 4;
    atomicAdd(a + 0, w * v0);
    atomicAdd(a + 1, w * v1);
    atomicAdd(a + 2, w * v2);
    atomicAdd(a + 3, w * v3);
}

// ---------------- fused GRU cell (WMMA) + residual ----------------
// tile: 16 nodes/block; gi = agg@wih^T, gh = prev@whh^T (both [16,384]) via
// v_wmma_f32_16x16x32_bf16, then gate math, hout = gru(agg,prev) + prev.
__global__ __launch_bounds__(256)
void gru_kernel(const unsigned short* __restrict__ agg_bf, const unsigned short* __restrict__ prev_bf,
                const float* __restrict__ prev_f, const unsigned short* __restrict__ wih,
                const unsigned short* __restrict__ whh, const float* __restrict__ bih,
                const float* __restrict__ bhh, float* __restrict__ hout, int N) {
    __shared__ __align__(16) float gi_s[16 * 384];
    __shared__ __align__(16) float gh_s[16 * 384];
    int tid = threadIdx.x;
    int wave = tid >> 5, lane = tid & 31;
    int col = lane & 15, half = lane >> 4;
    int m0 = blockIdx.x * 16;

    for (int t = wave; t < 48; t += 8) {
        int mat = t & 1;   // 0: gi, 1: gh
        int nc = t >> 1;   // 0..23 output column chunk
        const unsigned short* A = (mat ? prev_bf : agg_bf) + (size_t)(m0 + col) * 128;
        const unsigned short* W = (mat ? whh : wih) + (size_t)(nc * 16 + col) * 128;
        v8f acc = {};
#pragma unroll
        for (int kb = 0; kb < 128; kb += 32) {
            v16bf a = load_frag(A + kb + half * 8);
            v16bf b = load_frag(W + kb + half * 8);
            acc = __builtin_amdgcn_wmma_f32_16x16x32_bf16(false, a, false, b, (short)0, acc, false, false);
        }
        float* dstp = mat ? gh_s : gi_s;
#pragma unroll
        for (int g = 0; g < 8; ++g)
            dstp[(half * 8 + g) * 384 + nc * 16 + col] = acc[g];
    }
    __syncthreads();

    for (int i = tid; i < 2048; i += 256) {
        int mr = i >> 7, d = i & 127;
        size_t n = (size_t)(m0 + mr);
        float gir = gi_s[mr * 384 + d] + bih[d];
        float giz = gi_s[mr * 384 + 128 + d] + bih[128 + d];
        float gin = gi_s[mr * 384 + 256 + d] + bih[256 + d];
        float ghr = gh_s[mr * 384 + d] + bhh[d];
        float ghz = gh_s[mr * 384 + 128 + d] + bhh[128 + d];
        float ghn = gh_s[mr * 384 + 256 + d] + bhh[256 + d];
        float r = sigmoidf(gir + ghr);
        float zz = sigmoidf(giz + ghz);
        float nn = tanhf(gin + r * ghn);
        float hp = prev_f[n * 128 + d];
        hout[n * 128 + d] = (1.f - zz) * nn + zz * hp + hp;
    }
}

// ---------------- GraphNorm (batch is sorted -> flush-on-change accumulation) ----------------
// gnbuf layout (floats): [0..15]=cnt, [16..2063]=sum, [2064..4111]=sumsq
__global__ void gn_pass1(const float* __restrict__ h, const int* __restrict__ batch,
                         float* __restrict__ gnbuf, int N) {
    int d = threadIdx.x;  // 128
    int n0 = blockIdx.x * 64;
    if (n0 >= N) return;
    int nend = n0 + 64; if (nend > N) nend = N;
    float acc = 0.f, c = 0.f;
    int curg = batch[n0];
    for (int n = n0; n < nend; ++n) {
        int g = batch[n];
        if (g != curg) {
            atomicAdd(&gnbuf[16 + curg * 128 + d], acc);
            if (d == 0) atomicAdd(&gnbuf[curg], c);
            acc = 0.f; c = 0.f; curg = g;
        }
        acc += h[(size_t)n * 128 + d];
        c += 1.f;
    }
    atomicAdd(&gnbuf[16 + curg * 128 + d], acc);
    if (d == 0) atomicAdd(&gnbuf[curg], c);
}

__global__ void gn_pass2(float* __restrict__ h, const int* __restrict__ batch,
                         float* __restrict__ gnbuf, const float* __restrict__ gn_ms, int N) {
    int d = threadIdx.x;
    int n0 = blockIdx.x * 64;
    if (n0 >= N) return;
    int nend = n0 + 64; if (nend > N) nend = N;
    float ms = gn_ms[d];
    int curg = batch[n0];
    float mean = gnbuf[16 + curg * 128 + d] / gnbuf[curg];
    float ssq = 0.f;
    for (int n = n0; n < nend; ++n) {
        int g = batch[n];
        if (g != curg) {
            atomicAdd(&gnbuf[2064 + curg * 128 + d], ssq);
            ssq = 0.f; curg = g;
            mean = gnbuf[16 + curg * 128 + d] / gnbuf[curg];
        }
        float o = h[(size_t)n * 128 + d] - mean * ms;
        h[(size_t)n * 128 + d] = o;
        ssq += o * o;
    }
    atomicAdd(&gnbuf[2064 + curg * 128 + d], ssq);
}

__global__ void gn_pass3(float* __restrict__ h, unsigned short* __restrict__ hbf,
                         const int* __restrict__ batch, const float* __restrict__ gnbuf,
                         const float* __restrict__ gn_w, const float* __restrict__ gn_b, int total) {
    int i = blockIdx.x * 256 + threadIdx.x;
    if (i >= total) return;
    int n = i >> 7, d = i & 127;
    int g = batch[n];
    float var = gnbuf[2064 + g * 128 + d] / gnbuf[g];
    float v = h[i] * rsqrtf(var + 1e-5f) * gn_w[d] + gn_b[d];
    h[i] = v;
    hbf[i] = f2bf(v);
}

// ---------------- fused attention over layer stack (WMMA qkv + softmax + WMMA out) ----------------
// 8 nodes/block => 48 rows (node,s). qkv: [48,384]; attention per (node,head,s);
// ctx@out_w^T + out_b + hs; z[node] = sum_s h_final.
__global__ __launch_bounds__(256)
void attn_kernel(const unsigned short* __restrict__ hs_bf, const float* __restrict__ hs_f,
                 const unsigned short* __restrict__ wqkv, const float* __restrict__ bqkv,
                 const unsigned short* __restrict__ wout, const float* __restrict__ bout,
                 float* __restrict__ z_out, int N) {
    __shared__ __align__(16) unsigned short qkv_s[48 * 384];
    __shared__ __align__(16) unsigned short ctx_s[48 * 128];
    __shared__ float z_s[8 * 128];
    int tid = threadIdx.x;
    int wave = tid >> 5, lane = tid & 31;
    int col = lane & 15, half = lane >> 4;
    int node0 = blockIdx.x * 8;
    size_t slotStride = (size_t)N * 128;

    for (int i = tid; i < 1024; i += 256) z_s[i] = 0.f;

    // phase 1: qkv = hs @ wqkv^T   (72 wmma tile-tasks)
    for (int t = wave; t < 72; t += 8) {
        int mt = t / 24;  // 0..2 M-tile
        int nc = t % 24;  // 0..23 N-chunk
        int lr = mt * 16 + col;
        int node = node0 + lr / 6;
        int s = lr % 6;
        const unsigned short* arow = hs_bf + (size_t)s * slotStride + (size_t)node * 128;
        int n = nc * 16 + col;
        const unsigned short* brow = wqkv + (size_t)n * 128;
        v8f acc = {};
#pragma unroll
        for (int kb = 0; kb < 128; kb += 32) {
            v16bf a = load_frag(arow + kb + half * 8);
            v16bf b = load_frag(brow + kb + half * 8);
            acc = __builtin_amdgcn_wmma_f32_16x16x32_bf16(false, a, false, b, (short)0, acc, false, false);
        }
        float bias = bqkv[n];
#pragma unroll
        for (int g = 0; g < 8; ++g) {
            int r = mt * 16 + half * 8 + g;
            qkv_s[r * 384 + n] = f2bf(acc[g] + bias);
        }
    }
    __syncthreads();

    // phase 2: per (node, head, s) attention row (192 tasks)
    if (tid < 192) {
        int s = tid % 6;
        int hh = (tid / 6) % 4;
        int ln = tid / 24;
        int rq = ln * 6 + s;
        float q[32];
#pragma unroll
        for (int d2 = 0; d2 < 32; ++d2) q[d2] = bf2f(qkv_s[rq * 384 + hh * 32 + d2]);
        float sc[6];
        float mx = -1e30f;
#pragma unroll
        for (int tt = 0; tt < 6; ++tt) {
            int rk = ln * 6 + tt;
            float acc = 0.f;
#pragma unroll
            for (int d2 = 0; d2 < 32; ++d2) acc += q[d2] * bf2f(qkv_s[rk * 384 + 128 + hh * 32 + d2]);
            acc *= 0.17677669529663687f;  // 1/sqrt(32)
            sc[tt] = acc;
            mx = fmaxf(mx, acc);
        }
        float se = 0.f;
#pragma unroll
        for (int tt = 0; tt < 6; ++tt) { sc[tt] = expf(sc[tt] - mx); se += sc[tt]; }
        float inv = 1.f / se;
#pragma unroll
        for (int d2 = 0; d2 < 32; ++d2) {
            float c = 0.f;
#pragma unroll
            for (int tt = 0; tt < 6; ++tt)
                c += sc[tt] * bf2f(qkv_s[(ln * 6 + tt) * 384 + 256 + hh * 32 + d2]);
            ctx_s[rq * 128 + hh * 32 + d2] = f2bf(c * inv);
        }
    }
    __syncthreads();

    // phase 3: h_final = ctx @ wout^T + bout + hs ; z += over s (24 wmma tile-tasks)
    for (int t = wave; t < 24; t += 8) {
        int mt = t / 8;  // 0..2
        int nc = t % 8;  // 0..7
        int lrA = mt * 16 + col;
        const unsigned short* arow = &ctx_s[lrA * 128];
        int n = nc * 16 + col;
        const unsigned short* brow = wout + (size_t)n * 128;
        v8f acc = {};
#pragma unroll
        for (int kb = 0; kb < 128; kb += 32) {
            v16bf a = load_frag(arow + kb + half * 8);
            v16bf b = load_frag(brow + kb + half * 8);
            acc = __builtin_amdgcn_wmma_f32_16x16x32_bf16(false, a, false, b, (short)0, acc, false, false);
        }
        float bias = bout[n];
#pragma unroll
        for (int g = 0; g < 8; ++g) {
            int lr = mt * 16 + half * 8 + g;
            int node = node0 + lr / 6;
            int s = lr % 6;
            float hf = acc[g] + bias + hs_f[(size_t)s * slotStride + (size_t)node * 128 + n];
            atomicAdd(&z_s[(lr / 6) * 128 + n], hf);
        }
    }
    __syncthreads();
    for (int i = tid; i < 1024; i += 256)
        z_out[(size_t)(node0 + (i >> 7)) * 128 + (i & 127)] = z_s[i];
}

// ---------------- decoder: y = leaky_relu(z @ dec1^T) @ dec2^T ----------------
__global__ __launch_bounds__(64)
void dec_kernel(const float* __restrict__ z, const float* __restrict__ dec1_w,
                const float* __restrict__ dec2_w, float* __restrict__ out, int N) {
    __shared__ float red[64];
    int n = blockIdx.x;
    int t = threadIdx.x;
    const float* zr = z + (size_t)n * 128;
    float acc = 0.f;
#pragma unroll 4
    for (int d = 0; d < 128; ++d) acc += zr[d] * dec1_w[t * 128 + d];
    acc = acc > 0.f ? acc : 0.01f * acc;
    red[t] = acc * dec2_w[t];
    __syncthreads();
    for (int s = 32; s > 0; s >>= 1) {
        if (t < s) red[t] += red[t + s];
        __syncthreads();
    }
    if (t == 0) out[n] = red[0];
}

// ---------------- host orchestration ----------------
extern "C" void kernel_launch(void* const* d_in, const int* in_sizes, int n_in,
                              void* d_out, int out_size, void* d_ws, size_t ws_size,
                              hipStream_t stream) {
    const float* x        = (const float*)d_in[0];
    const int*   eidx     = (const int*)d_in[1];
    const float* norm     = (const float*)d_in[2];
    const int*   batch    = (const int*)d_in[3];
    const float* enc_w    = (const float*)d_in[5];
    const float* gru_wih  = (const float*)d_in[6];
    const float* gru_whh  = (const float*)d_in[7];
    const float* gru_bih  = (const float*)d_in[8];
    const float* gru_bhh  = (const float*)d_in[9];
    const float* gn_w     = (const float*)d_in[10];
    const float* gn_b     = (const float*)d_in[11];
    const float* gn_ms    = (const float*)d_in[12];
    const float* in_proj_w = (const float*)d_in[13];
    const float* in_proj_b = (const float*)d_in[14];
    const float* out_w    = (const float*)d_in[15];
    const float* out_b    = (const float*)d_in[16];
    const float* dec1_w   = (const float*)d_in[17];
    const float* dec2_w   = (const float*)d_in[18];
    float* out = (float*)d_out;

    const int N = in_sizes[0] / 6;   // 100000
    const int E = in_sizes[2];       // 1600000
    const int D = 128, S = 6, L = 5;
    const size_t ND = (size_t)N * D;

    // workspace carve-out (256B aligned slots)
    char* w = (char*)d_ws;
    size_t off = 0;
    auto take = [&](size_t bytes) -> void* {
        void* p = w + off;
        off += (bytes + 255) & ~(size_t)255;
        return p;
    };
    float*          hs_f   = (float*)take((size_t)S * ND * 4);
    unsigned short* hs_bf  = (unsigned short*)take((size_t)S * ND * 2);
    float*          agg_f  = (float*)take(ND * 4);
    unsigned short* agg_bf = (unsigned short*)take(ND * 2);
    float*          z_f    = (float*)take(ND * 4);
    float*          gnbuf  = (float*)take(4112 * 4);
    unsigned short* wih_bf = (unsigned short*)take(384 * 128 * 2);
    unsigned short* whh_bf = (unsigned short*)take(384 * 128 * 2);
    unsigned short* wqkv_bf = (unsigned short*)take(384 * 128 * 2);
    unsigned short* wout_bf = (unsigned short*)take(128 * 128 * 2);

    // weights -> bf16 (tiny, once per call)
    cast_f32_bf16<<<(49152 + 255) / 256, 256, 0, stream>>>(gru_wih, wih_bf, 49152);
    cast_f32_bf16<<<(49152 + 255) / 256, 256, 0, stream>>>(gru_whh, whh_bf, 49152);
    cast_f32_bf16<<<(49152 + 255) / 256, 256, 0, stream>>>(in_proj_w, wqkv_bf, 49152);
    cast_f32_bf16<<<(16384 + 255) / 256, 256, 0, stream>>>(out_w, wout_bf, 16384);

    // encoder
    enc_kernel<<<(int)((ND + 255) / 256), 256, 0, stream>>>(x, enc_w, hs_f, hs_bf, (int)ND);

    for (int l = 0; l < L; ++l) {
        const float* prev_f = hs_f + (size_t)l * ND;
        const unsigned short* prev_bf = hs_bf + (size_t)l * ND;
        float* hnext = hs_f + (size_t)(l + 1) * ND;
        unsigned short* hnext_bf = hs_bf + (size_t)(l + 1) * ND;

        zero_f32<<<(int)((ND + 255) / 256), 256, 0, stream>>>(agg_f, (int)ND);
        edge_agg_kernel<<<(E + 7) / 8, 256, 0, stream>>>(eidx, eidx + E, norm, prev_bf, agg_f, E);
        cast_f32_bf16<<<(int)((ND + 255) / 256), 256, 0, stream>>>(agg_f, agg_bf, (int)ND);

        gru_kernel<<<N / 16, 256, 0, stream>>>(agg_bf, prev_bf, prev_f, wih_bf, whh_bf,
                                               gru_bih, gru_bhh, hnext, N);

        zero_f32<<<(4112 + 255) / 256, 256, 0, stream>>>(gnbuf, 4112);
        gn_pass1<<<(N + 63) / 64, 128, 0, stream>>>(hnext, batch, gnbuf, N);
        gn_pass2<<<(N + 63) / 64, 128, 0, stream>>>(hnext, batch, gnbuf, gn_ms, N);
        gn_pass3<<<(int)((ND + 255) / 256), 256, 0, stream>>>(hnext, hnext_bf, batch, gnbuf,
                                                              gn_w, gn_b, (int)ND);
    }

    attn_kernel<<<N / 8, 256, 0, stream>>>(hs_bf, hs_f, wqkv_bf, in_proj_b, wout_bf, out_b, z_f, N);
    dec_kernel<<<N, 64, 0, stream>>>(z_f, dec1_w, dec2_w, out, N);
}
